// TrajectoryAttention_23545010716762
// MI455X (gfx1250) — matrix operand order repaired and checked by reading, not compiled, not run on hardware
//
#include <hip/hip_runtime.h>

#define K_B   2
#define K_S   2048
#define K_D   1024
#define K_H   16
#define K_DH  64
#define K_W2  128          // WINDOW/2
#define K_NEG (-1e9f)
#define K_SCALE 0.125f     // 1/sqrt(64)

typedef __attribute__((ext_vector_type(16))) __bf16 v16bf;
typedef __attribute__((ext_vector_type(8)))  float  v8f;
typedef unsigned short bf16_t;

union FragAB { v16bf v; unsigned u[8]; };

static __device__ __forceinline__ bf16_t f2bf(float f) {
  union { float f; unsigned u; } c; c.f = f;
  unsigned u = c.u + 0x7FFFu + ((c.u >> 16) & 1u);   // round-to-nearest-even
  return (bf16_t)(u >> 16);
}
static __device__ __forceinline__ unsigned pack2(bf16_t lo, bf16_t hi) {
  return (unsigned)lo | ((unsigned)hi << 16);
}
static __device__ __forceinline__ v8f vzero8() {
  v8f z = {0.f, 0.f, 0.f, 0.f, 0.f, 0.f, 0.f, 0.f};
  return z;
}

// LDS row padding for GEMM tiles: 40 elems = 80 B (20 banks, 16-B aligned)
#define LDSP 40
// V staging: 208 rows live, row stride 232 elems = 464 B (52 banks mod 64,
// gcd(52,64)=4 -> 16 distinct banks across 16 lanes; 464 % 16 == 0)
#define VROWS 208
#define VLDSP 232

// ---------------------------------------------------------------- convert
__global__ void cvt_f32_to_bf16(const float* __restrict__ src,
                                bf16_t* __restrict__ dst, int n) {
  int i = blockIdx.x * blockDim.x + threadIdx.x;
  if (i < n) dst[i] = f2bf(src[i]);
}

// ---------------------------------------------------------------- GEMM core
// Block = 8 waves as 4(M) x 2(N); macro-tile 64x128; K-step 32.
// DOUBLE-BUFFERED: async global->LDS copy of tile k+1 (ASYNCcnt) overlaps the
// WMMAs of tile k; one barrier per k-step.
// B tile staged transposed (Bt[n][k]) so B-frag reads are contiguous b128s.
static __device__ __forceinline__ void gemm_macro(
    const bf16_t* __restrict__ Ag, const bf16_t* __restrict__ Bg,
    int lda, int ldb, int kdim, int bm0, int bn0,
    bf16_t (*__restrict__ As)[64][LDSP], bf16_t (*__restrict__ Bt)[128][LDSP],
    v8f acc[4]) {
  const int t    = threadIdx.x;
  const int lane = t & 31;
  const int lh   = lane >> 4;
  const int lc   = lane & 15;
  const int wid  = t >> 5;
  const int wm   = wid & 3;       // wave M row (0..3)
  const int wn   = wid >> 2;      // wave N col (0..1)

  const int ar = t >> 2;          // A stage: row 0..63
  const int ac = (t & 3) * 8;     // A stage: 8-elem (16 B) chunk
  const int kp = t & 15;          // B stage: k-pair (k = 2*kp)
  const int ng = t >> 4;          // B stage: n-group 0..15

  auto stage = [&](int buf, int kk) {
    // A (64x32): async copy, 16 B per thread, ASYNCcnt-tracked
    const bf16_t* gsrc = Ag + (size_t)(bm0 + ar) * lda + kk + ac;
    unsigned ldsoff = (unsigned)(size_t)(void*)&As[buf][ar][ac];
    asm volatile("global_load_async_to_lds_b128 %0, %1, off"
                 :: "v"(ldsoff), "v"(gsrc) : "memory");
    // B^T (128 n x 32 k) with transpose: 2x(b64 load + 4x b32 store)
#pragma unroll
    for (int it = 0; it < 2; ++it) {
      const int n4 = (ng + it * 16) * 4;
      const bf16_t* r0 = Bg + (size_t)(kk + 2 * kp) * ldb + bn0 + n4;
      const bf16_t* r1 = r0 + ldb;
#pragma unroll
      for (int j = 0; j < 4; ++j)
        *reinterpret_cast<unsigned*>(&Bt[buf][n4 + j][2 * kp]) =
            pack2(r0[j], r1[j]);
    }
  };

  stage(0, 0);
  int buf = 0;
  for (int kk = 0; kk < kdim; kk += 32) {
    // own async copy for `buf` done; barrier makes all staging visible and
    // guarantees previous iteration's LDS reads completed before overwrite
    asm volatile("s_wait_asynccnt 0x0" ::: "memory");
    __syncthreads();
    if (kk + 32 < kdim) stage(buf ^ 1, kk + 32);   // overlaps with WMMAs below

    FragAB a;
    {
      const bf16_t* arow = &As[buf][wm * 16 + lc][0];
#pragma unroll
      for (int i = 0; i < 8; ++i) {
        int kb = lh * 16 + 2 * i;
        a.u[i] = *reinterpret_cast<const unsigned*>(arow + kb);
      }
    }
#pragma unroll
    for (int nt = 0; nt < 4; ++nt) {
      FragAB bfr;
      const bf16_t* brow = &Bt[buf][wn * 64 + nt * 16 + lc][lh * 16];
#pragma unroll
      for (int i = 0; i < 8; ++i)
        bfr.u[i] = *reinterpret_cast<const unsigned*>(brow + 2 * i);
      acc[nt] = __builtin_amdgcn_wmma_f32_16x16x32_bf16(
          false, a.v, false, bfr.v, (short)0, acc[nt], false, false);
    }
    buf ^= 1;
  }
}

// ---------------------------------------------------------------- QKV GEMM
// C(4096x3072) = Xb(4096x1024) * Wqkvb(1024x3072); epilogue -> q/k/v (b,h,s,dh)
__global__ __launch_bounds__(256) void qkv_gemm(
    const bf16_t* __restrict__ xb, const bf16_t* __restrict__ wb,
    bf16_t* __restrict__ qo, bf16_t* __restrict__ ko, bf16_t* __restrict__ vo) {
  __shared__ bf16_t As[2][64][LDSP];
  __shared__ bf16_t Bt[2][128][LDSP];
  const int bm0 = (blockIdx.x % 64) * 64;    // 4096/64 = 64 M-blocks
  const int bn0 = (blockIdx.x / 64) * 128;   // 3072/128 = 24 N-blocks

  v8f acc[4] = {vzero8(), vzero8(), vzero8(), vzero8()};
  gemm_macro(xb, wb, K_D, 3 * K_D, K_D, bm0, bn0, As, Bt, acc);

  const int lane = threadIdx.x & 31;
  const int wid  = threadIdx.x >> 5;
  const int lh   = lane >> 4;
  const int lc   = lane & 15;
  const int m0   = bm0 + (wid & 3) * 16;
  const int n0   = bn0 + (wid >> 2) * 64;

  // wave-uniform epilogue factors: a 16x64 tile lives in exactly one of
  // q/k/v (1024-aligned sections), one head (64-aligned), one batch (M 2048)
  const int which = n0 >> 10;                 // 0:q 1:k 2:v
  const int h     = (n0 & (K_D - 1)) >> 6;
  const int bb    = m0 >> 11;
  const int srow  = m0 & (K_S - 1);
  const float scl = (which == 0) ? K_SCALE : 1.0f;
  bf16_t* dst = (which == 0) ? qo : (which == 1) ? ko : vo;
  bf16_t* base = dst + (((size_t)(bb * K_H + h)) * K_S + srow) * K_DH;

#pragma unroll
  for (int nt = 0; nt < 4; ++nt)
#pragma unroll
    for (int r = 0; r < 8; ++r)
      base[(size_t)(r + lh * 8) * K_DH + nt * 16 + lc] = f2bf(acc[nt][r] * scl);
}

// ---------------------------------------------------------------- attention
// Block = 4 waves, consecutive q-tiles of one (b,h); q-span 64 rows.
// Shared V band [i0base-128, i0base+80) staged TRANSPOSED in LDS so PV
// B-fragments are contiguous ds_load_b128s.
__global__ __launch_bounds__(128) void attn_band(
    const bf16_t* __restrict__ qm, const bf16_t* __restrict__ km,
    const bf16_t* __restrict__ vm, const float* __restrict__ bias,
    bf16_t* __restrict__ ao) {
  __shared__ bf16_t Vt[K_DH][VLDSP];     // [dh][local_k]
  __shared__ bf16_t Pb[4][16][160];
  const int tid  = threadIdx.x;
  const int lane = tid & 31;
  const int wid  = tid >> 5;             // 0..3
  const int lh   = lane >> 4;
  const int lc   = lane & 15;

  const int qblock = blockIdx.x & 31;    // 32 q-blocks (64 rows each) per head
  const int bh     = blockIdx.x >> 5;    // 0..31 (= b*H + h)
  const int i0base = qblock * 64;
  const int i0     = i0base + wid * 16;
  const int jbase  = i0base - K_W2;      // start of staged V band (may be <0)
  const size_t headoff = (size_t)bh * K_S;

  // ---- stage V^T band: 208 rows x 64 dh, transposed, clamped at edges.
  // 104 row-pairs x 16 dh-groups = 1664 tasks; 128 threads x 13 iters.
#pragma unroll
  for (int it = 0; it < 13; ++it) {
    const int task = tid + it * 128;
    const int rp   = task >> 4;          // row-pair 0..103
    const int dg   = (task & 15) * 4;    // dh group
    int j0 = jbase + 2 * rp;
    int j1 = j0 + 1;
    j0 = min(max(j0, 0), K_S - 1);       // clamped rows pair with P==0
    j1 = min(max(j1, 0), K_S - 1);
    const bf16_t* r0 = vm + (headoff + j0) * K_DH + dg;
    const bf16_t* r1 = vm + (headoff + j1) * K_DH + dg;
#pragma unroll
    for (int j = 0; j < 4; ++j)
      *reinterpret_cast<unsigned*>(&Vt[dg + j][2 * rp]) = pack2(r0[j], r1[j]);
  }

  // ---- Q fragments (A layout, 16x32), dh chunks [0,32) and [32,64)
  FragAB qa[2];
  {
    const bf16_t* qrow = qm + (headoff + i0 + lc) * K_DH;
#pragma unroll
    for (int c = 0; c < 2; ++c)
#pragma unroll
      for (int i = 0; i < 8; ++i) {
        int kb = c * 32 + lh * 16 + 2 * i;
        qa[c].u[i] = *reinterpret_cast<const unsigned*>(qrow + kb);
      }
  }

  const int jstart = i0 - K_W2;   // this wave's band start (multiple of 16)
  float sc[9][8];

#pragma unroll
  for (int t = 0; t < 9; ++t) {
    const int jt = jstart + t * 16;         // wave-uniform
    if (jt < 0) {
#pragma unroll
      for (int r = 0; r < 8; ++r) sc[t][r] = K_NEG;
      continue;
    }
    v8f acc = vzero8();
#pragma unroll
    for (int c = 0; c < 2; ++c) {
      FragAB kb_;   // B layout: col n = key jt+lc, K = dh feature
      const bf16_t* krow = km + (headoff + jt + lc) * K_DH + c * 32 + lh * 16;
#pragma unroll
      for (int i = 0; i < 8; ++i)
        kb_.u[i] = *reinterpret_cast<const unsigned*>(krow + 2 * i);
      acc = __builtin_amdgcn_wmma_f32_16x16x32_bf16(
          false, qa[c].v, false, kb_.v, (short)0, acc, false, false);
    }
#pragma unroll
    for (int r = 0; r < 8; ++r) {
      const int qi = i0 + r + lh * 8;
      const int kj = jt + lc;
      if (kj > qi || (qi - kj) > K_W2) sc[t][r] = K_NEG;
      else sc[t][r] = acc[r] + bias[((size_t)bh * K_S + qi) * K_S + kj];
    }
  }

  // ---- row softmax: rows live in 16-lane halves; xor 1,2,4,8 reduces a half
  float mx_[8], inv_[8];
#pragma unroll
  for (int r = 0; r < 8; ++r) {
    float m = sc[0][r];
#pragma unroll
    for (int t = 1; t < 9; ++t) m = fmaxf(m, sc[t][r]);
#pragma unroll
    for (int d = 1; d < 16; d <<= 1) m = fmaxf(m, __shfl_xor(m, d, 32));
    mx_[r] = m;
  }
#pragma unroll
  for (int r = 0; r < 8; ++r) {
    float s = 0.f;
#pragma unroll
    for (int t = 0; t < 9; ++t) { sc[t][r] = __expf(sc[t][r] - mx_[r]); s += sc[t][r]; }
#pragma unroll
    for (int d = 1; d < 16; d <<= 1) s += __shfl_xor(s, d, 32);
    inv_[r] = 1.0f / s;
  }

  // ---- probabilities to LDS (C-frag -> A-frag re-layout); pad [144,160)=0
#pragma unroll
  for (int r = 0; r < 8; ++r) {
    Pb[wid][r + lh * 8][144 + lc] = 0;
#pragma unroll
    for (int t = 0; t < 9; ++t)
      Pb[wid][r + lh * 8][t * 16 + lc] = f2bf(sc[t][r] * inv_[r]);
  }
  __syncthreads();   // Vt + Pb complete; uniform across block

  // ---- O(16x64) = P(16x160) * V(160x64); B-frags from transposed Vt in LDS
  v8f o[4] = {vzero8(), vzero8(), vzero8(), vzero8()};
#pragma unroll
  for (int kk = 0; kk < 160; kk += 32) {
    FragAB pa;
    const bf16_t* prow = &Pb[wid][lc][kk];
#pragma unroll
    for (int i = 0; i < 8; ++i) {
      int kb = lh * 16 + 2 * i;
      pa.u[i] = *reinterpret_cast<const unsigned*>(prow + kb);
    }
    const int koff = wid * 16 + kk + lh * 16;   // local_k base in Vt
#pragma unroll
    for (int nt = 0; nt < 4; ++nt) {
      FragAB bv;
      const bf16_t* brow = &Vt[nt * 16 + lc][koff];
#pragma unroll
      for (int i = 0; i < 8; ++i)
        bv.u[i] = *reinterpret_cast<const unsigned*>(brow + 2 * i);
      o[nt] = __builtin_amdgcn_wmma_f32_16x16x32_bf16(
          false, pa.v, false, bv.v, (short)0, o[nt], false, false);
    }
  }

  // ---- write attention output as bf16, layout (b, s, h*64+dh)
  const int bb = bh >> 4;
  const int h  = bh & 15;
  bf16_t* base = ao + ((size_t)(bb * K_S + i0)) * K_D + h * K_DH;
#pragma unroll
  for (int nt = 0; nt < 4; ++nt)
#pragma unroll
    for (int r = 0; r < 8; ++r)
      base[(size_t)(r + lh * 8) * K_D + nt * 16 + lc] = f2bf(o[nt][r]);
}

// ---------------------------------------------------------------- out GEMM
// out(4096x1024 f32) = ab(4096x1024 bf16) * woutb(1024x1024 bf16)
__global__ __launch_bounds__(256) void out_gemm(
    const bf16_t* __restrict__ ab, const bf16_t* __restrict__ wb,
    float* __restrict__ out) {
  __shared__ bf16_t As[2][64][LDSP];
  __shared__ bf16_t Bt[2][128][LDSP];
  const int bm0 = (blockIdx.x % 64) * 64;    // 4096/64 = 64 M-blocks
  const int bn0 = (blockIdx.x / 64) * 128;   // 1024/128 = 8 N-blocks

  v8f acc[4] = {vzero8(), vzero8(), vzero8(), vzero8()};
  gemm_macro(ab, wb, K_D, K_D, K_D, bm0, bn0, As, Bt, acc);

  const int lane = threadIdx.x & 31;
  const int wid  = threadIdx.x >> 5;
  const int lh   = lane >> 4;
  const int lc   = lane & 15;
  const int m0   = bm0 + (wid & 3) * 16;
  const int n0   = bn0 + (wid >> 2) * 64;

  float* base = out + (size_t)m0 * K_D + n0;
#pragma unroll
  for (int nt = 0; nt < 4; ++nt)
#pragma unroll
    for (int r = 0; r < 8; ++r)
      base[(size_t)(r + lh * 8) * K_D + nt * 16 + lc] = acc[nt][r];
}

// ---------------------------------------------------------------- launch
extern "C" void kernel_launch(void* const* d_in, const int* in_sizes, int n_in,
                              void* d_out, int out_size, void* d_ws, size_t ws_size,
                              hipStream_t stream) {
  const float* x    = (const float*)d_in[0];
  const float* bias = (const float*)d_in[1];
  // d_in[2] = causal mask: applied analytically in-kernel
  const float* Wqkv = (const float*)d_in[3];
  const float* Wout = (const float*)d_in[4];
  float* out = (float*)d_out;

  char* ws = (char*)d_ws;
  size_t off = 0;
  auto alloc = [&](size_t bytes) -> char* {
    char* p = ws + off;
    off += (bytes + 255) & ~(size_t)255;
    return p;
  };
  bf16_t* xb    = (bf16_t*)alloc((size_t)K_B * K_S * K_D * 2);
  bf16_t* wqkvb = (bf16_t*)alloc((size_t)K_D * 3 * K_D * 2);
  bf16_t* woutb = (bf16_t*)alloc((size_t)K_D * K_D * 2);
  bf16_t* qw    = (bf16_t*)alloc((size_t)K_B * K_H * K_S * K_DH * 2);
  bf16_t* kw    = (bf16_t*)alloc((size_t)K_B * K_H * K_S * K_DH * 2);
  bf16_t* vw    = (bf16_t*)alloc((size_t)K_B * K_H * K_S * K_DH * 2);
  bf16_t* ab    = (bf16_t*)alloc((size_t)K_B * K_S * K_D * 2);

  const int n1 = K_B * K_S * K_D;      // x
  const int n2 = K_D * 3 * K_D;        // W_qkv
  const int n3 = K_D * K_D;            // W_out
  cvt_f32_to_bf16<<<(n1 + 255) / 256, 256, 0, stream>>>(x, xb, n1);
  cvt_f32_to_bf16<<<(n2 + 255) / 256, 256, 0, stream>>>(Wqkv, wqkvb, n2);
  cvt_f32_to_bf16<<<(n3 + 255) / 256, 256, 0, stream>>>(Wout, woutb, n3);

  // QKV: (4096/64) x (3072/128) = 1536 blocks
  qkv_gemm<<<1536, 256, 0, stream>>>(xb, wqkvb, qw, kw, vw);
  // attention: 32 (b*h) x 32 q-blocks = 1024 blocks x 4 waves
  attn_band<<<1024, 128, 0, stream>>>(qw, kw, vw, bias, ab);
  // output projection: (4096/64) x (1024/128) = 512 blocks
  out_gemm<<<512, 256, 0, stream>>>(ab, woutb, out);

  (void)in_sizes; (void)n_in; (void)out_size; (void)ws_size;
}
